// Head_20023137534487
// MI455X (gfx1250) — matrix operand order repaired and checked by reading, not compiled, not run on hardware
//
#include <hip/hip_runtime.h>
#include <hip/hip_bf16.h>
#include <stdint.h>

// Problem constants (from reference): B=8, T=2048, C=1024 (embed), H=64 (head)
#define B_DIM 8
#define T_DIM 2048
#define C_DIM 1024
#define H_DIM 64
#define QK_SCALE 0.03125f   // C ** -0.5 = 1/32 (reference scales by embed dim)

// Async-to-LDS staging of K/V chunks (CDNA5 ASYNCcnt path). Inline asm is the
// toolchain-portable route (the clang builtin arity differs across toolchains).
#define USE_ASYNC_STAGING 1

#if defined(__has_builtin)
#if __has_builtin(__builtin_amdgcn_sched_barrier)
#define SCHED_BARRIER() __builtin_amdgcn_sched_barrier(0)
#endif
#endif
#ifndef SCHED_BARRIER
#define SCHED_BARRIER() do {} while (0)
#endif

typedef __attribute__((ext_vector_type(16))) __bf16 v16bf;
typedef __attribute__((ext_vector_type(8)))  float  v8f;

union Frag {
    v16bf    v;
    uint32_t u[8];
    uint4    q[2];
};

__device__ __forceinline__ uint16_t f2bf(float f) {
    union { float f; uint32_t u; } x; x.f = f;
    uint32_t r = 0x7FFFu + ((x.u >> 16) & 1u);   // round-to-nearest-even
    return (uint16_t)((x.u + r) >> 16);
}
__device__ __forceinline__ uint32_t pack2(float a, float b) {
    return (uint32_t)f2bf(a) | ((uint32_t)f2bf(b) << 16);
}
__device__ __forceinline__ v8f vzero() {
    v8f z;
    #pragma unroll
    for (int i = 0; i < 8; ++i) z[i] = 0.0f;
    return z;
}
__device__ __forceinline__ v8f wmma_bf16(const Frag& a, const Frag& b, v8f c) {
    return __builtin_amdgcn_wmma_f32_16x16x32_bf16(
        /*neg_a=*/false, a.v, /*neg_b=*/false, b.v,
        /*c_mod=*/(short)0, c, /*reuse_a=*/false, /*reuse_b=*/false);
}

// ---------------------------------------------------------------------------
// Kernel 0: transpose + convert Wq/Wk/Wv (C x H fp32) -> (H x C bf16)
// ---------------------------------------------------------------------------
__global__ void __launch_bounds__(256)
wconv_kernel(const float* __restrict__ Wq, const float* __restrict__ Wk,
             const float* __restrict__ Wv,
             uint16_t* __restrict__ Wtq, uint16_t* __restrict__ Wtk,
             uint16_t* __restrict__ Wtv) {
    int idx = blockIdx.x * blockDim.x + threadIdx.x;   // 3 * 65536 total
    int mat = idx >> 16;
    int e   = idx & 0xFFFF;
    int k   = e >> 6;      // 0..1023
    int h   = e & 63;      // 0..63
    const float* src = (mat == 0) ? Wq : (mat == 1) ? Wk : Wv;
    uint16_t*    dst = (mat == 0) ? Wtq : (mat == 1) ? Wtk : Wtv;
    dst[h * C_DIM + k] = f2bf(src[k * H_DIM + h]);
}

// ---------------------------------------------------------------------------
// Kernel 1: Q/K/V projections. One wave per 16-row tile of x.
//   All 12 B-fragments load before a sched_barrier so the 24 b128 loads issue
//   as clauses with staggered partial loadcnt waits instead of 12x wait(0).
// Outputs: Qbf,Kbf row-major (B,T,H) bf16; Vt transposed (B,H,T) bf16.
// ---------------------------------------------------------------------------
__global__ void __launch_bounds__(128)
proj_kernel(const float* __restrict__ x,
            const uint16_t* __restrict__ Wtq, const uint16_t* __restrict__ Wtk,
            const uint16_t* __restrict__ Wtv,
            uint16_t* __restrict__ Qbf, uint16_t* __restrict__ Kbf,
            uint16_t* __restrict__ Vt) {
    const int lane = threadIdx.x & 31;
    const int wave = threadIdx.x >> 5;
    const int half = lane >> 4;     // 0: lanes 0-15, 1: lanes 16-31
    const int lm   = lane & 15;
    const int tile = blockIdx.x * 4 + wave;        // 0 .. B*T/16 - 1
    const int b    = tile >> 7;                    // 128 tiles per batch
    const int t0   = (tile & 127) << 4;

    v8f accq[4], acck[4], accv[4];
    #pragma unroll
    for (int n = 0; n < 4; ++n) { accq[n] = vzero(); acck[n] = vzero(); accv[n] = vzero(); }

    const float* xrow = x + ((size_t)(b * T_DIM + t0 + lm)) * C_DIM;

    for (int kc = 0; kc < C_DIM; kc += 32) {
        if (kc + 32 < C_DIM)
            __builtin_prefetch(xrow + kc + 32, 0, 0);

        // A-fragment (16x32 bf16), contiguous k order, same as B:
        //  lanes 0-15: K{kc+0..15}; lanes 16-31: K{kc+16..31}
        Frag a;
        const float* p0 = xrow + kc + half * 16;
        const float* p1 = xrow + kc + 8 + half * 16;
        float4 f0 = *(const float4*)(p0);
        float4 f1 = *(const float4*)(p0 + 4);
        float4 f2 = *(const float4*)(p1);
        float4 f3 = *(const float4*)(p1 + 4);
        a.u[0] = pack2(f0.x, f0.y); a.u[1] = pack2(f0.z, f0.w);
        a.u[2] = pack2(f1.x, f1.y); a.u[3] = pack2(f1.z, f1.w);
        a.u[4] = pack2(f2.x, f2.y); a.u[5] = pack2(f2.z, f2.w);
        a.u[6] = pack2(f3.x, f3.y); a.u[7] = pack2(f3.z, f3.w);

        // Load ALL B-fragments first; barrier keeps loads from sinking into
        // the WMMA sequence (otherwise the scheduler serializes load->wait(0)
        // ->wmma on one recycled register block).
        Frag bq[4], bk[4], bv[4];
        #pragma unroll
        for (int nt = 0; nt < 4; ++nt) {
            const size_t base = (size_t)(nt * 16 + lm) * C_DIM + kc + 16 * half;
            bq[nt].q[0] = *(const uint4*)(Wtq + base);
            bq[nt].q[1] = *(const uint4*)(Wtq + base + 8);
            bk[nt].q[0] = *(const uint4*)(Wtk + base);
            bk[nt].q[1] = *(const uint4*)(Wtk + base + 8);
            bv[nt].q[0] = *(const uint4*)(Wtv + base);
            bv[nt].q[1] = *(const uint4*)(Wtv + base + 8);
        }
        SCHED_BARRIER();
        #pragma unroll
        for (int nt = 0; nt < 4; ++nt) {
            accq[nt] = wmma_bf16(a, bq[nt], accq[nt]);
            acck[nt] = wmma_bf16(a, bk[nt], acck[nt]);
            accv[nt] = wmma_bf16(a, bv[nt], accv[nt]);
        }
    }

    // Epilogue: C/D layout -> memory. lane holds column h; VGPR r is row r+8*half.
    #pragma unroll
    for (int nt = 0; nt < 4; ++nt) {
        const int h = nt * 16 + lm;
        #pragma unroll
        for (int r = 0; r < 8; ++r) {
            const int row = t0 + r + 8 * half;
            const size_t qi = ((size_t)(b * T_DIM) + row) * H_DIM + h;
            Qbf[qi] = f2bf(accq[nt][r]);
            Kbf[qi] = f2bf(acck[nt][r]);
            Vt[((size_t)(b * H_DIM + h)) * T_DIM + row] = f2bf(accv[nt][r]);
        }
    }
}

// ---------------------------------------------------------------------------
// Kernel 2: causal flash attention. One wave per 16-query tile, 32-key chunks.
// K/V chunks are double-buffered in LDS via async-to-LDS copies (ASYNCcnt):
// issue next chunk's 16 async b128 copies, wait asynccnt<=16 for the current
// chunk (async loads complete in order), compute from LDS via ds_read_b128.
// ---------------------------------------------------------------------------
__global__ void __launch_bounds__(128)
attn_kernel(const uint16_t* __restrict__ Qbf, const uint16_t* __restrict__ Kbf,
            const uint16_t* __restrict__ Vt, float* __restrict__ out) {
    const int lane = threadIdx.x & 31;
    const int wave = threadIdx.x >> 5;
    const int half = lane >> 4;
    const int lm   = lane & 15;
    const int tile = blockIdx.x * 4 + wave;
    const int b    = tile >> 7;
    const int qt   = (tile & 127) << 4;

    __shared__ __align__(16) uint16_t Kst[4][2][32 * 64];  // 32 keys x 64 h
    __shared__ __align__(16) uint16_t Vst[4][2][64 * 32];  // 64 h x 32 keys
    __shared__ __align__(16) uint16_t Pst[4][16 * 32];     // P staging
    uint16_t* P = Pst[wave];

    const uint16_t* Kbase = Kbf + (size_t)b * T_DIM * H_DIM;
    const uint16_t* Vbase = Vt  + (size_t)b * H_DIM * T_DIM;

    // Stage one 32-key chunk (K: 4KB, V: 4KB) with 16 async b128 copies/wave.
    auto stage_chunk = [&](int ks, int buf) {
#if USE_ASYNC_STAGING
        #pragma unroll
        for (int j = 0; j < 8; ++j) {
            const int idx = lane + 32 * j;
            const int row = idx >> 3, seg = idx & 7;    // K rows: 128B = 8 segs
            const uint16_t* g = Kbase + (size_t)(ks + row) * H_DIM + seg * 8;
            unsigned l = (unsigned)(uintptr_t)&Kst[wave][buf][row * 64 + seg * 8];
            asm volatile("global_load_async_to_lds_b128 %0, %1, off"
                         :: "v"(l), "v"(g) : "memory");
        }
        #pragma unroll
        for (int j = 0; j < 8; ++j) {
            const int idx = lane + 32 * j;
            const int row = idx >> 2, seg = idx & 3;    // V rows: 64B = 4 segs
            const uint16_t* g = Vbase + (size_t)row * T_DIM + ks + seg * 8;
            unsigned l = (unsigned)(uintptr_t)&Vst[wave][buf][row * 32 + seg * 8];
            asm volatile("global_load_async_to_lds_b128 %0, %1, off"
                         :: "v"(l), "v"(g) : "memory");
        }
#else
        #pragma unroll
        for (int j = 0; j < 8; ++j) {
            const int idx = lane + 32 * j;
            const int row = idx >> 3, seg = idx & 7;
            *(uint4*)&Kst[wave][buf][row * 64 + seg * 8] =
                *(const uint4*)(Kbase + (size_t)(ks + row) * H_DIM + seg * 8);
        }
        #pragma unroll
        for (int j = 0; j < 8; ++j) {
            const int idx = lane + 32 * j;
            const int row = idx >> 2, seg = idx & 3;
            *(uint4*)&Vst[wave][buf][row * 32 + seg * 8] =
                *(const uint4*)(Vbase + (size_t)row * T_DIM + ks + seg * 8);
        }
#endif
    };

    // Q A-fragments for the two K-steps over H (H 0..31 and 32..63)
    const uint16_t* qrow = Qbf + ((size_t)(b * T_DIM) + qt + lm) * H_DIM;
    Frag aq0, aq1;
    aq0.q[0] = *(const uint4*)(qrow + 16 * half);
    aq0.q[1] = *(const uint4*)(qrow + 8 + 16 * half);
    aq1.q[0] = *(const uint4*)(qrow + 32 + 16 * half);
    aq1.q[1] = *(const uint4*)(qrow + 40 + 16 * half);

    float m_i[8], l_i[8];
    v8f o[4];
    #pragma unroll
    for (int r = 0; r < 8; ++r) { m_i[r] = -1.0e30f; l_i[r] = 0.0f; }
    #pragma unroll
    for (int n = 0; n < 4; ++n) o[n] = vzero();

    const int kend = qt + 16;                      // causal bound (exclusive)
    stage_chunk(0, 0);

    for (int ks = 0; ks < kend; ks += 32) {
        const int buf = (ks >> 5) & 1;
        if (ks + 32 < kend) {
            stage_chunk(ks + 32, buf ^ 1);         // prefetch next chunk
#if USE_ASYNC_STAGING
            asm volatile("s_wait_asynccnt 0x10" ::: "memory");  // current done
#endif
        } else {
#if USE_ASYNC_STAGING
            asm volatile("s_wait_asynccnt 0x0" ::: "memory");
#endif
        }
        const uint16_t* Kc = Kst[wave][buf];
        const uint16_t* Vc = Vst[wave][buf];

        // ---- S = Q * K^T for 16 queries x 32 keys (two 16x16 tiles) ----
        v8f s[2];
        #pragma unroll
        for (int t = 0; t < 2; ++t) {
            const uint16_t* krow = Kc + (t * 16 + lm) * 64;   // lane's key row
            Frag b0, b1;
            b0.q[0] = *(const uint4*)(krow + 16 * half);
            b0.q[1] = *(const uint4*)(krow + 16 * half + 8);
            b1.q[0] = *(const uint4*)(krow + 32 + 16 * half);
            b1.q[1] = *(const uint4*)(krow + 32 + 16 * half + 8);
            v8f c = vzero();
            c = wmma_bf16(aq0, b0, c);
            c = wmma_bf16(aq1, b1, c);
            s[t] = c;
        }

        // ---- scale + causal mask + online softmax ----
        float pv0[8], pv1[8], cm[8];
        #pragma unroll
        for (int r = 0; r < 8; ++r) {
            const int qm = qt + r + 8 * half;      // this VGPR's query row
            float v0 = s[0][r] * QK_SCALE;
            float v1 = s[1][r] * QK_SCALE;
            if (ks + lm      > qm) v0 = -1.0e30f;
            if (ks + 16 + lm > qm) v1 = -1.0e30f;
            pv0[r] = v0; pv1[r] = v1;
            cm[r] = fmaxf(v0, v1);
        }
        #pragma unroll
        for (int msk = 1; msk < 16; msk <<= 1) {
            #pragma unroll
            for (int r = 0; r < 8; ++r)
                cm[r] = fmaxf(cm[r], __shfl_xor(cm[r], msk, 32));
        }
        float sf[8], rs[8];
        #pragma unroll
        for (int r = 0; r < 8; ++r) {
            const float mn = fmaxf(m_i[r], cm[r]);
            sf[r]  = __expf(m_i[r] - mn);
            m_i[r] = mn;
            const float p0 = __expf(pv0[r] - mn);
            const float p1 = __expf(pv1[r] - mn);
            P[(r + 8 * half) * 32 + lm]      = f2bf(p0);
            P[(r + 8 * half) * 32 + 16 + lm] = f2bf(p1);
            rs[r] = p0 + p1;
        }
        #pragma unroll
        for (int msk = 1; msk < 16; msk <<= 1) {
            #pragma unroll
            for (int r = 0; r < 8; ++r)
                rs[r] += __shfl_xor(rs[r], msk, 32);
        }
        #pragma unroll
        for (int r = 0; r < 8; ++r) l_i[r] = l_i[r] * sf[r] + rs[r];
        #pragma unroll
        for (int nt = 0; nt < 4; ++nt)
            #pragma unroll
            for (int r = 0; r < 8; ++r) o[nt][r] *= sf[r];

        // ---- P (C-layout fp32) -> A-layout bf16 via LDS ----
        Frag ap;
        const uint16_t* prow = P + lm * 32;        // A row m = lm
        ap.q[0] = *(const uint4*)(prow + 16 * half);
        ap.q[1] = *(const uint4*)(prow + 8 + 16 * half);

        // ---- O += P * V  (V h-major in LDS: key dim contiguous) ----
        Frag bv[4];
        #pragma unroll
        for (int nt = 0; nt < 4; ++nt) {
            const uint16_t* vrow = Vc + (nt * 16 + lm) * 32 + 16 * half;
            bv[nt].q[0] = *(const uint4*)(vrow);
            bv[nt].q[1] = *(const uint4*)(vrow + 8);
        }
        SCHED_BARRIER();
        #pragma unroll
        for (int nt = 0; nt < 4; ++nt)
            o[nt] = wmma_bf16(ap, bv[nt], o[nt]);
    }

    // ---- epilogue: normalize by row sums and store fp32 output ----
    #pragma unroll
    for (int nt = 0; nt < 4; ++nt) {
        const int h = nt * 16 + lm;
        #pragma unroll
        for (int r = 0; r < 8; ++r) {
            const int row = qt + r + 8 * half;
            out[((size_t)(b * T_DIM) + row) * H_DIM + h] = o[nt][r] / l_i[r];
        }
    }
}

// ---------------------------------------------------------------------------
extern "C" void kernel_launch(void* const* d_in, const int* in_sizes, int n_in,
                              void* d_out, int out_size, void* d_ws, size_t ws_size,
                              hipStream_t stream) {
    const float* x  = (const float*)d_in[0];
    const float* Wq = (const float*)d_in[1];
    const float* Wk = (const float*)d_in[2];
    const float* Wv = (const float*)d_in[3];
    float* out = (float*)d_out;

    // Workspace layout (bf16 elements): 3 transposed weights + Q + K + Vt
    uint16_t* Wtq = (uint16_t*)d_ws;
    uint16_t* Wtk = Wtq + (size_t)H_DIM * C_DIM;
    uint16_t* Wtv = Wtk + (size_t)H_DIM * C_DIM;
    uint16_t* Qbf = Wtv + (size_t)H_DIM * C_DIM;
    uint16_t* Kbf = Qbf + (size_t)B_DIM * T_DIM * H_DIM;
    uint16_t* Vt  = Kbf + (size_t)B_DIM * T_DIM * H_DIM;
    // total: (3*64K + 3*1M) bf16 ≈ 6.7 MB of d_ws

    wconv_kernel<<<(3 * H_DIM * C_DIM) / 256, 256, 0, stream>>>(Wq, Wk, Wv, Wtq, Wtk, Wtv);

    const int tiles = B_DIM * T_DIM / 16;          // 1024 row tiles, 4 waves/block
    proj_kernel<<<tiles / 4, 128, 0, stream>>>(x, Wtq, Wtk, Wtv, Qbf, Kbf, Vt);
    attn_kernel<<<tiles / 4, 128, 0, stream>>>(Qbf, Kbf, Vt, out);
}
